// GraphClusterNet_77644418777366
// MI455X (gfx1250) — matrix operand order repaired and checked by reading, not compiled, not run on hardware
//
#include <hip/hip_runtime.h>
#include <cstdint>

typedef __attribute__((ext_vector_type(16))) __bf16 v16bf;
typedef __attribute__((ext_vector_type(8)))  float  v8f;

#define NN 50000
#define NE 800000

// ---------------- order-preserving float<->uint for atomic segment-max ----------------
static __device__ __forceinline__ unsigned enc_ord(float f) {
  unsigned u = __float_as_uint(f);
  return (u & 0x80000000u) ? ~u : (u | 0x80000000u);
}
static __device__ __forceinline__ float dec_ord(unsigned u) {
  unsigned b = (u & 0x80000000u) ? (u & 0x7fffffffu) : ~u;
  return __uint_as_float(b);
}

// ---------------- CDNA5 async global->LDS copy (16B per lane, ASYNCcnt) ----------------
static __device__ __forceinline__ void async_copy16(void* lds_dst, const void* gsrc) {
  unsigned ldsa = (unsigned)(uintptr_t)lds_dst;           // low 32 bits = LDS byte offset
  unsigned long long ga = (unsigned long long)(uintptr_t)gsrc;
  asm volatile("global_load_async_to_lds_b128 %0, %1, off"
               :: "v"(ldsa), "v"(ga) : "memory");
}
static __device__ __forceinline__ void wait_async0() {
  asm volatile("s_wait_asynccnt 0" ::: "memory");
}

// ---------------- generic zero fill ----------------
__global__ void fill_zero_u32(unsigned* __restrict__ p, size_t n) {
  size_t i = (size_t)blockIdx.x * blockDim.x + threadIdx.x;
  if (i < n) p[i] = 0u;
}

// ---------------- one-shot converts: x -> bf16, W[KxN] f32 -> Wt[NxK] bf16 ------------
__global__ __launch_bounds__(256) void f32_to_bf16(const float* __restrict__ s,
                                                   __bf16* __restrict__ d, int n) {
  int i = blockIdx.x * blockDim.x + threadIdx.x;
  if (i < n) d[i] = (__bf16)s[i];
}
__global__ __launch_bounds__(256) void w_to_bf16_t(const float* __restrict__ W,
                                                   __bf16* __restrict__ Wt, int K) {
  int i = blockIdx.x * blockDim.x + threadIdx.x;  // over K*128, coalesced in n
  if (i >= K * 128) return;
  int k = i >> 7, n = i & 127;
  Wt[(size_t)n * K + k] = (__bf16)W[i];
}

// ---------------- WMMA GEMM: C[M x 128] = A_bf16[M x K] @ Wt_bf16[128 x K]^T (+bias) --
// block = 256 threads = 8 waves; wave computes a 16 x 128 strip (8 accumulators).
// Tiles staged into LDS with global_load_async_to_lds_b128 (no VALU in the copy path).
template <int K>
__global__ __launch_bounds__(256) void gemm_bf16_n128(const __bf16* __restrict__ A,
                                                      const __bf16* __restrict__ Wt,
                                                      const float* __restrict__ bias,
                                                      float* __restrict__ C, int M) {
  constexpr int KT = 32;
  __shared__ __bf16 As[128][40];   // 80B row stride: 16B-aligned chunks, 20-bank stagger
  __shared__ __bf16 Bs[128][40];   // Bs[n][k]

  const int tid  = threadIdx.x;
  const int wave = tid >> 5;
  const int lane = tid & 31;
  const int row0 = blockIdx.x * 128;

  v8f zero = {};
  v8f acc[8];
#pragma unroll
  for (int i = 0; i < 8; ++i) acc[i] = zero;

  for (int k0 = 0; k0 < K; k0 += KT) {
    // 512 16B chunks per tile; 2 A-chunks + 2 B-chunks per thread, all async
#pragma unroll
    for (int c = 0; c < 2; ++c) {
      int chunk = tid * 2 + c;          // 0..511
      int r = chunk >> 2, sub = chunk & 3;
      int gr = row0 + r;
      if (gr < M)                        // stale LDS rows only feed unstored outputs
        async_copy16(&As[r][sub * 8], A + (size_t)gr * K + k0 + sub * 8);
      async_copy16(&Bs[r][sub * 8], Wt + (size_t)r * K + k0 + sub * 8);
    }
    if (k0 + KT < K)
      __builtin_prefetch((const void*)(A + (size_t)(row0 + (tid >> 1)) * K + k0 + KT), 0, 1);
    wait_async0();
    __syncthreads();

    // A fragment: lane = M row (lanes 16-31 carry K halves 8..15 / 24..31)
    v16bf a;
    const int arow  = wave * 16 + (lane & 15);
    const int khalf = (lane >> 4) * 8;
#pragma unroll
    for (int t = 0; t < 8; ++t) {
      int kk = (t >> 2) * 16 + khalf + (t & 3) * 2;
      a[2 * t]     = As[arow][kk];
      a[2 * t + 1] = As[arow][kk + 1];
    }
    // B fragments: lane -> N = lane&15, same K order as A (khalf + (t>>2)*16 + (t&3)*2)
    const int kb = (lane >> 4) * 16;
#pragma unroll
    for (int nt = 0; nt < 8; ++nt) {
      v16bf b;
      const int bn = nt * 16 + (lane & 15);
#pragma unroll
      for (int t = 0; t < 8; ++t) {
        b[2 * t]     = Bs[bn][(t >> 2) * 16 + khalf + (t & 3) * 2];
        b[2 * t + 1] = Bs[bn][(t >> 2) * 16 + khalf + (t & 3) * 2 + 1];
      }
      acc[nt] = __builtin_amdgcn_wmma_f32_16x16x32_bf16(
          false, a, false, b, (short)0, acc[nt], false, false);
    }
    __syncthreads();
  }

  // epilogue: C/D layout -> lane N = lane&15, VGPR r -> M = r + 8*(lane>=16)
  const int colb   = lane & 15;
  const int rowoff = (lane >> 4) * 8;
#pragma unroll
  for (int nt = 0; nt < 8; ++nt) {
    int col  = nt * 16 + colb;
    float bv = bias ? bias[col] : 0.f;
#pragma unroll
    for (int r = 0; r < 8; ++r) {
      int gr = row0 + wave * 16 + rowoff + r;
      if (gr < M) C[(size_t)gr * 128 + col] = acc[nt][r] + bv;
    }
  }
}

// ---------------- PReLU + LayerNorm (wave per node), emits f32 + bf16 ----------------
__global__ __launch_bounds__(256) void prelu_ln(const float* __restrict__ hin,
                                                const float* __restrict__ pa,
                                                const float* __restrict__ g,
                                                const float* __restrict__ b,
                                                float* __restrict__ hf,
                                                __bf16* __restrict__ hb, int N) {
  int wid  = (blockIdx.x * blockDim.x + threadIdx.x) >> 5;
  int lane = threadIdx.x & 31;
  if (wid >= N) return;
  float a = pa[0];
  int c0 = lane * 4;
  float4 v = *(const float4*)(hin + (size_t)wid * 128 + c0);
  v.x = v.x >= 0.f ? v.x : a * v.x;
  v.y = v.y >= 0.f ? v.y : a * v.y;
  v.z = v.z >= 0.f ? v.z : a * v.z;
  v.w = v.w >= 0.f ? v.w : a * v.w;
  float s = v.x + v.y + v.z + v.w;
  for (int o = 16; o; o >>= 1) s += __shfl_xor(s, o);
  float mean = s * (1.f / 128.f);
  float dx = v.x - mean, dy = v.y - mean, dz = v.z - mean, dw = v.w - mean;
  float q = dx * dx + dy * dy + dz * dz + dw * dw;
  for (int o = 16; o; o >>= 1) q += __shfl_xor(q, o);
  float rstd = rsqrtf(q * (1.f / 128.f) + 1e-5f);
  float4 gv = *(const float4*)(g + c0);
  float4 bv = *(const float4*)(b + c0);
  float o0 = dx * rstd * gv.x + bv.x;
  float o1 = dy * rstd * gv.y + bv.y;
  float o2 = dz * rstd * gv.z + bv.z;
  float o3 = dw * rstd * gv.w + bv.w;
  float*  fo = hf + (size_t)wid * 128 + c0;
  __bf16* bo = hb + (size_t)wid * 128 + c0;
  fo[0] = o0; fo[1] = o1; fo[2] = o2; fo[3] = o3;
  bo[0] = (__bf16)o0; bo[1] = (__bf16)o1; bo[2] = (__bf16)o2; bo[3] = (__bf16)o3;
}

// ---------------- GAT attention scores es/ed (wave per node) ----------------
__global__ __launch_bounds__(256) void gat_attscore(const float* __restrict__ xh,
                                                    const float* __restrict__ asrc,
                                                    const float* __restrict__ adst,
                                                    float* __restrict__ es,
                                                    float* __restrict__ ed, int N, int heads) {
  int wid  = (blockIdx.x * blockDim.x + threadIdx.x) >> 5;
  int lane = threadIdx.x & 31;
  if (wid >= N) return;
  int c0 = lane * 4;
  float4 xv = *(const float4*)(xh + (size_t)wid * 128 + c0);
  float4 av = *(const float4*)(asrc + c0);
  float4 dv = *(const float4*)(adst + c0);
  float ss = xv.x * av.x + xv.y * av.y + xv.z * av.z + xv.w * av.w;
  float sd = xv.x * dv.x + xv.y * dv.y + xv.z * dv.z + xv.w * dv.w;
  if (heads == 1) { ss += __shfl_xor(ss, 16); sd += __shfl_xor(sd, 16); }
  for (int o = 8; o; o >>= 1) { ss += __shfl_xor(ss, o); sd += __shfl_xor(sd, o); }
  if (lane == 0) { es[(size_t)wid * heads] = ss; ed[(size_t)wid * heads] = sd; }
  if (heads == 2 && lane == 16) { es[(size_t)wid * 2 + 1] = ss; ed[(size_t)wid * 2 + 1] = sd; }
}

// ---------------- edge pass 1: segment max (thread per edge, incl. self loops) -------
__global__ void gat_max(const int* __restrict__ ei, const float* __restrict__ es,
                        const float* __restrict__ ed, unsigned* __restrict__ menc,
                        int E, int N, int heads) {
  int t = blockIdx.x * blockDim.x + threadIdx.x;
  if (t >= E + N) return;
  int s, d;
  if (t < E) { s = ei[t]; d = ei[E + t]; } else { s = d = t - E; }
  for (int h = 0; h < heads; ++h) {
    float e = es[(size_t)s * heads + h] + ed[(size_t)d * heads + h];
    e = e >= 0.f ? e : 0.2f * e;
    atomicMax(&menc[(size_t)d * heads + h], enc_ord(e));
  }
}

// ---------------- edge pass 2: p=exp(e-m), den += p, acc += p * xh[src] (wave/edge) --
__global__ __launch_bounds__(256) void gat_acc(const int* __restrict__ ei,
                                               const float* __restrict__ es,
                                               const float* __restrict__ ed,
                                               const unsigned* __restrict__ menc,
                                               const float* __restrict__ xh,
                                               float* __restrict__ den,
                                               float* __restrict__ acc,
                                               int E, int N, int heads) {
  int wid  = (blockIdx.x * blockDim.x + threadIdx.x) >> 5;
  int lane = threadIdx.x & 31;
  if (wid >= E + N) return;
  int s, d;
  if (wid < E) { s = ei[wid]; d = ei[E + wid]; } else { s = d = wid - E; }
  float e0 = es[(size_t)s * heads] + ed[(size_t)d * heads];
  e0 = e0 >= 0.f ? e0 : 0.2f * e0;
  float p0 = __expf(e0 - dec_ord(menc[(size_t)d * heads]));
  float p1 = 0.f;
  if (heads == 2) {
    float e1 = es[(size_t)s * 2 + 1] + ed[(size_t)d * 2 + 1];
    e1 = e1 >= 0.f ? e1 : 0.2f * e1;
    p1 = __expf(e1 - dec_ord(menc[(size_t)d * 2 + 1]));
  }
  if (lane == 0) atomicAdd(&den[(size_t)d * heads], p0);
  if (heads == 2 && lane == 1) atomicAdd(&den[(size_t)d * 2 + 1], p1);
  int c0 = lane * 4;
  float w = (heads == 2 && lane >= 16) ? p1 : p0;  // head = col/64
  float4 xv = *(const float4*)(xh + (size_t)s * 128 + c0);
  float* ap = acc + (size_t)d * 128 + c0;
  atomicAdd(ap + 0, w * xv.x);
  atomicAdd(ap + 1, w * xv.y);
  atomicAdd(ap + 2, w * xv.z);
  atomicAdd(ap + 3, w * xv.w);
}

// ---------------- finalize: y = acc/den + bias + idn; per-column BN partials ---------
__global__ __launch_bounds__(128) void gat_final_stats(const float* __restrict__ acc,
                                                       const float* __restrict__ den,
                                                       const float* __restrict__ gb,
                                                       const float* __restrict__ idn,
                                                       float* __restrict__ y,
                                                       float* __restrict__ colsum,
                                                       float* __restrict__ colsq,
                                                       int N, int heads, int rpb) {
  int c = threadIdx.x;
  int h = (heads == 2) ? (c >> 6) : 0;
  int r0 = blockIdx.x * rpb;
  int r1 = r0 + rpb; if (r1 > N) r1 = N;
  float s = 0.f, s2 = 0.f;
  float bias = gb[c];
  for (int r = r0; r < r1; ++r) {
    float v = acc[(size_t)r * 128 + c] / den[(size_t)r * heads + h] + bias
            + idn[(size_t)r * 128 + c];
    y[(size_t)r * 128 + c] = v;
    s += v; s2 += v * v;
  }
  atomicAdd(&colsum[c], s);
  atomicAdd(&colsq[c], s2);
}

__global__ void bn_params(const float* __restrict__ cs, const float* __restrict__ cq,
                          const float* __restrict__ g, const float* __restrict__ b,
                          float* __restrict__ scale, float* __restrict__ shift, float invN) {
  int c = threadIdx.x;
  float m  = cs[c] * invN;
  float v  = cq[c] * invN - m * m;
  float rs = rsqrtf(v + 1e-5f);
  float sc = g[c] * rs;
  scale[c] = sc;
  shift[c] = b[c] - m * sc;
}

__global__ __launch_bounds__(256) void bn_apply(const float* __restrict__ y,
                                                const float* __restrict__ scale,
                                                const float* __restrict__ shift,
                                                float* __restrict__ hf,
                                                __bf16* __restrict__ hb, int total) {
  int i = blockIdx.x * blockDim.x + threadIdx.x;
  if (i >= total) return;
  int c = i & 127;
  float v = y[i] * scale[c] + shift[c];
  v = v > 0.f ? v : (__expf(v) - 1.f);  // ELU(alpha=1)
  if (hf) hf[i] = v;
  hb[i] = (__bf16)v;
}

// ---------------- cluster K/V projection: 16x128, trivial ----------------
__global__ void proto_kv(const float* __restrict__ protos,
                         const float* __restrict__ Wk, const float* __restrict__ bk,
                         const float* __restrict__ Wv, const float* __restrict__ bv,
                         float* __restrict__ kb, float* __restrict__ vb) {
  int i = blockIdx.x * blockDim.x + threadIdx.x;
  if (i >= 16 * 128) return;
  int r = i >> 7, c = i & 127;
  float sk = bk[c], sv = bv[c];
  for (int t = 0; t < 128; ++t) {
    float pv = protos[r * 128 + t];
    sk += pv * Wk[t * 128 + c];
    sv += pv * Wv[t * 128 + c];
  }
  kb[i] = sk; vb[i] = sv;
}

// ---------------- cluster attention (wave per node): softmax over 16, attn@v ---------
__global__ __launch_bounds__(256) void cluster_attn(const float* __restrict__ q,
                                                    const float* __restrict__ kb,
                                                    const float* __restrict__ vb,
                                                    __bf16* __restrict__ cf, int N) {
  __shared__ float ks[16 * 128];
  __shared__ float vs[16 * 128];
  for (int i = threadIdx.x; i < 16 * 128; i += 256) { ks[i] = kb[i]; vs[i] = vb[i]; }
  __syncthreads();
  int wid  = (blockIdx.x * blockDim.x + threadIdx.x) >> 5;
  int lane = threadIdx.x & 31;
  if (wid >= N) return;
  const float* qr = q + (size_t)wid * 128;
  int j = lane & 15, half = lane >> 4;
  float s = 0.f;
  for (int t = 0; t < 64; ++t) { int c = half * 64 + t; s += qr[c] * ks[j * 128 + c]; }
  s += __shfl_xor(s, 16);
  s *= 0.08838834764831845f;  // 1/sqrt(128)
  float mx = s;
  for (int o = 8; o; o >>= 1) mx = fmaxf(mx, __shfl_xor(mx, o));
  float p = __expf(s - mx);
  float dn = p;
  for (int o = 8; o; o >>= 1) dn += __shfl_xor(dn, o);
  p /= dn;
  int c0 = lane * 4;
  float o0 = 0.f, o1 = 0.f, o2 = 0.f, o3 = 0.f;
  for (int jj = 0; jj < 16; ++jj) {
    float pj = __shfl(p, jj);
    const float* vr = vs + jj * 128 + c0;
    o0 += pj * vr[0]; o1 += pj * vr[1]; o2 += pj * vr[2]; o3 += pj * vr[3];
  }
  __bf16* o = cf + (size_t)wid * 128 + c0;
  o[0] = (__bf16)o0; o[1] = (__bf16)o1; o[2] = (__bf16)o2; o[3] = (__bf16)o3;
}

// ---------------- comb = [h2 | cluster_feat] as bf16 ----------------
__global__ __launch_bounds__(256) void build_comb(const __bf16* __restrict__ h2b,
                                                  const float* __restrict__ cfeat,
                                                  __bf16* __restrict__ comb, int N) {
  int i = blockIdx.x * blockDim.x + threadIdx.x;
  if (i >= N * 256) return;
  int r = i >> 8, c = i & 255;
  comb[i] = (c < 128) ? h2b[(size_t)r * 128 + c] : (__bf16)cfeat[(size_t)r * 128 + (c - 128)];
}

// ---------------- final: leaky_relu(z) @ cW2 + cb2 (wave per node) ----------------
__global__ __launch_bounds__(256) void final_out(const float* __restrict__ z,
                                                 const float* __restrict__ w,
                                                 const float* __restrict__ b,
                                                 float* __restrict__ out, int N) {
  int wid  = (blockIdx.x * blockDim.x + threadIdx.x) >> 5;
  int lane = threadIdx.x & 31;
  if (wid >= N) return;
  int c0 = lane * 4;
  float4 zv = *(const float4*)(z + (size_t)wid * 128 + c0);
  float4 wv = *(const float4*)(w + c0);
  zv.x = zv.x >= 0.f ? zv.x : 0.2f * zv.x;
  zv.y = zv.y >= 0.f ? zv.y : 0.2f * zv.y;
  zv.z = zv.z >= 0.f ? zv.z : 0.2f * zv.z;
  zv.w = zv.w >= 0.f ? zv.w : 0.2f * zv.w;
  float s = zv.x * wv.x + zv.y * wv.y + zv.z * wv.z + zv.w * wv.w;
  for (int o = 16; o; o >>= 1) s += __shfl_xor(s, o);
  if (lane == 0) out[wid] = s + b[0];
}

// ======================================================================================
extern "C" void kernel_launch(void* const* d_in, const int* in_sizes, int n_in,
                              void* d_out, int out_size, void* d_ws, size_t ws_size,
                              hipStream_t stream) {
  (void)in_sizes; (void)n_in; (void)out_size; (void)ws_size;
  const float* x       = (const float*)d_in[0];
  const int*   ei      = (const int*)d_in[1];
  const float* enc_W   = (const float*)d_in[2];
  const float* enc_b   = (const float*)d_in[3];
  const float* prelu_a = (const float*)d_in[4];
  const float* ln_g    = (const float*)d_in[5];
  const float* ln_b    = (const float*)d_in[6];
  const float* g1_W    = (const float*)d_in[7];
  const float* g1_as   = (const float*)d_in[8];
  const float* g1_ad   = (const float*)d_in[9];
  const float* g1_b    = (const float*)d_in[10];
  const float* bn1_g   = (const float*)d_in[11];
  const float* bn1_b   = (const float*)d_in[12];
  const float* g2_W    = (const float*)d_in[13];
  const float* g2_as   = (const float*)d_in[14];
  const float* g2_ad   = (const float*)d_in[15];
  const float* g2_b    = (const float*)d_in[16];
  const float* bn2_g   = (const float*)d_in[17];
  const float* bn2_b   = (const float*)d_in[18];
  const float* protos  = (const float*)d_in[19];
  const float* Wq = (const float*)d_in[20]; const float* bq = (const float*)d_in[21];
  const float* Wk = (const float*)d_in[22]; const float* bk = (const float*)d_in[23];
  const float* Wv = (const float*)d_in[24]; const float* bv = (const float*)d_in[25];
  const float* Wo = (const float*)d_in[26]; const float* bo = (const float*)d_in[27];
  const float* cW1 = (const float*)d_in[28]; const float* cb1 = (const float*)d_in[29];
  const float* cW2 = (const float*)d_in[30]; const float* cb2 = (const float*)d_in[31];

  const int N = NN, E = NE;
  const size_t NF = (size_t)N * 128;

  char* base = (char*)d_ws;
  size_t off = 0;
  auto alloc = [&](size_t bytes) -> char* {
    char* p = base + off;
    off += (bytes + 255) & ~(size_t)255;
    return p;
  };
  float*  P0   = (float*)alloc(NF * 4);       // h_enc -> h1(f32)
  float*  P1   = (float*)alloc(NF * 4);       // h_ln(f32, idn1) -> z_raw
  float*  P2   = (float*)alloc(NF * 4);       // xh -> q -> cluster_feat
  float*  P3   = (float*)alloc(NF * 4);       // acc / y (re-zeroed per layer)
  __bf16* BF0  = (__bf16*)alloc(NF * 2);      // h_ln -> h1 -> h2 (bf16 GEMM inputs)
  __bf16* BF1  = (__bf16*)alloc(NF * 2);      // attn@v (bf16)
  __bf16* BF2  = (__bf16*)alloc(NF * 4);      // x_bf16, later comb [N x 256] bf16
  float*  esb  = (float*)alloc((size_t)N * 2 * 4);
  float*  edb  = (float*)alloc((size_t)N * 2 * 4);
  unsigned* menc = (unsigned*)alloc((size_t)N * 2 * 4);
  float*  denb = (float*)alloc((size_t)N * 2 * 4);
  float*  colsum = (float*)alloc(128 * 4);
  float*  colsq  = (float*)alloc(128 * 4);
  float*  bscale = (float*)alloc(128 * 4);
  float*  bshift = (float*)alloc(128 * 4);
  float*  kbuf = (float*)alloc(16 * 128 * 4);
  float*  vbuf = (float*)alloc(16 * 128 * 4);
  __bf16* WtE = (__bf16*)alloc(128 * 256 * 2);  // transposed bf16 weights
  __bf16* Wt1 = (__bf16*)alloc(128 * 128 * 2);
  __bf16* Wt2 = (__bf16*)alloc(128 * 128 * 2);
  __bf16* WtQ = (__bf16*)alloc(128 * 128 * 2);
  __bf16* WtO = (__bf16*)alloc(128 * 128 * 2);
  __bf16* WtC = (__bf16*)alloc(128 * 256 * 2);

  const int GEMM_GRID = (N + 127) / 128;
  const int nodeWaveBlocks = (N * 32 + 255) / 256;
  const int elemBlocks = (int)((NF + 255) / 256);
  const int ET = E + N;

  // 0) one-shot packs: x -> bf16 (in BF2), weights -> transposed bf16
  f32_to_bf16<<<(N * 256 + 255) / 256, 256, 0, stream>>>(x, BF2, N * 256);
  w_to_bf16_t<<<128, 256, 0, stream>>>(enc_W, WtE, 256);
  w_to_bf16_t<<<64, 256, 0, stream>>>(g1_W, Wt1, 128);
  w_to_bf16_t<<<64, 256, 0, stream>>>(g2_W, Wt2, 128);
  w_to_bf16_t<<<64, 256, 0, stream>>>(Wq, WtQ, 128);
  w_to_bf16_t<<<64, 256, 0, stream>>>(Wo, WtO, 128);
  w_to_bf16_t<<<128, 256, 0, stream>>>(cW1, WtC, 256);

  // 1) encoder GEMM (K=256) + bias
  gemm_bf16_n128<256><<<GEMM_GRID, 256, 0, stream>>>(BF2, WtE, enc_b, P0, N);
  // 2) PReLU + LayerNorm -> h_ln f32 (P1) + bf16 (BF0)
  prelu_ln<<<nodeWaveBlocks, 256, 0, stream>>>(P0, prelu_a, ln_g, ln_b, P1, BF0, N);

  auto gat_layer = [&](const __bf16* hin_b, const __bf16* Wt, const float* as_,
                       const float* ad_, const float* gb, const float* bng,
                       const float* bnb, const float* idn, int heads,
                       float* hf_out, __bf16* hb_out) {
    gemm_bf16_n128<128><<<GEMM_GRID, 256, 0, stream>>>(hin_b, Wt, nullptr, P2, N);
    gat_attscore<<<nodeWaveBlocks, 256, 0, stream>>>(P2, as_, ad_, esb, edb, N, heads);
    fill_zero_u32<<<(N * heads + 255) / 256, 256, 0, stream>>>(menc, (size_t)N * heads);
    fill_zero_u32<<<(N * heads + 255) / 256, 256, 0, stream>>>((unsigned*)denb, (size_t)N * heads);
    fill_zero_u32<<<elemBlocks, 256, 0, stream>>>((unsigned*)P3, NF);
    fill_zero_u32<<<1, 256, 0, stream>>>((unsigned*)colsum, 128);
    fill_zero_u32<<<1, 256, 0, stream>>>((unsigned*)colsq, 128);
    gat_max<<<(ET + 255) / 256, 256, 0, stream>>>(ei, esb, edb, menc, E, N, heads);
    gat_acc<<<(int)(((size_t)ET * 32 + 255) / 256), 256, 0, stream>>>(
        ei, esb, edb, menc, P2, denb, P3, E, N, heads);
    gat_final_stats<<<(N + 63) / 64, 128, 0, stream>>>(P3, denb, gb, idn, P3, colsum, colsq,
                                                       N, heads, 64);
    bn_params<<<1, 128, 0, stream>>>(colsum, colsq, bng, bnb, bscale, bshift, 1.0f / (float)N);
    bn_apply<<<elemBlocks, 256, 0, stream>>>(P3, bscale, bshift, hf_out, hb_out, (int)NF);
  };

  // 3) GAT layer 1 (heads=2), idn = h_ln(P1) -> h1: f32 in P0, bf16 in BF0
  gat_layer(BF0, Wt1, g1_as, g1_ad, g1_b, bn1_g, bn1_b, P1, 2, P0, BF0);
  // 4) GAT layer 2 (heads=1), idn = h1(P0) -> h2: bf16 in BF0
  gat_layer(BF0, Wt2, g2_as, g2_ad, g2_b, bn2_g, bn2_b, P0, 1, nullptr, BF0);

  // 5) cluster attention
  proto_kv<<<(16 * 128 + 255) / 256, 256, 0, stream>>>(protos, Wk, bk, Wv, bv, kbuf, vbuf);
  gemm_bf16_n128<128><<<GEMM_GRID, 256, 0, stream>>>(BF0, WtQ, bq, P2, N);   // q
  cluster_attn<<<nodeWaveBlocks, 256, 0, stream>>>(P2, kbuf, vbuf, BF1, N);  // attn@v
  gemm_bf16_n128<128><<<GEMM_GRID, 256, 0, stream>>>(BF1, WtO, bo, P2, N);   // cluster_feat

  // 6) classifier
  build_comb<<<(N * 256 + 255) / 256, 256, 0, stream>>>(BF0, P2, BF2, N);    // BF2: x dead
  gemm_bf16_n128<256><<<GEMM_GRID, 256, 0, stream>>>(BF2, WtC, cb1, P1, N);  // z_raw
  final_out<<<nodeWaveBlocks, 256, 0, stream>>>(P1, cW2, cb2, (float*)d_out, N);
}
